// Transformer_20383914787558
// MI455X (gfx1250) — matrix operand order repaired and checked by reading, not compiled, not run on hardware
//
#include <hip/hip_runtime.h>
#include <hip/hip_bf16.h>

typedef __attribute__((ext_vector_type(16))) _Float16 v16h;
typedef __attribute__((ext_vector_type(8)))  _Float16 v8h;
typedef __attribute__((ext_vector_type(2)))  _Float16 v2h;
typedef __attribute__((ext_vector_type(8)))  float    v8f;
typedef int v4i __attribute__((vector_size(16)));

namespace {
constexpr int kB  = 8;
constexpr int kT  = 2048;
constexpr int kC  = 96;     // d_model
constexpr int kH  = 3;      // heads
constexpr int kDh = 32;     // head dim
constexpr int kFF = 2048;   // d_ff
constexpr int kTilesT = kT / 16;   // 128
constexpr float kScale = 0.17677669529663687f; // 1/sqrt(32)
constexpr float kEps = 1e-5f;
}

// ---------- optional CDNA5 async global->LDS copy path ----------
#if defined(__has_builtin)
#if __has_builtin(__builtin_amdgcn_global_load_async_to_lds_b128)
#define ASYNC_LDS 1
#endif
#if __has_builtin(__builtin_amdgcn_s_wait_asynccnt)
#define HAVE_WAIT_ASYNC_BUILTIN 1
#endif
#endif

__device__ inline void copy16(_Float16* dst_lds, const _Float16* src_glb) {
#ifdef ASYNC_LDS
  __builtin_amdgcn_global_load_async_to_lds_b128(
      (__attribute__((address_space(1))) v4i*)src_glb,
      (__attribute__((address_space(3))) v4i*)dst_lds, 0, 0);
#else
  *(v8h*)dst_lds = *(const v8h*)src_glb;
#endif
}

__device__ inline void wait_async() {
#ifdef ASYNC_LDS
#ifdef HAVE_WAIT_ASYNC_BUILTIN
  __builtin_amdgcn_s_wait_asynccnt(0);
#else
  asm volatile("s_wait_asynccnt 0" ::: "memory");
#endif
#endif
}

__device__ inline v8f wmma16(v16h a, v16h b, v8f c) {
  return __builtin_amdgcn_wmma_f32_16x16x32_f16(false, a, false, b, (short)0, c,
                                                false, false);
}

// A fragment from row-major LDS [16][ldr]; two aligned 16B loads per lane.
__device__ inline v16h ldsA(const _Float16* base, int ldr, int lane) {
  int m = lane & 15, hs = lane >> 4;
  const _Float16* p = base + m * ldr + hs * 16;
  v8h lo = *(const v8h*)p;
  v8h hi = *(const v8h*)(p + 8);
  return __builtin_shufflevector(lo, hi, 0, 1, 2, 3, 4, 5, 6, 7,
                                 8, 9, 10, 11, 12, 13, 14, 15);
}

// B fragment from swizzled LDS: element B[k][n] at (k>>4)*256 + n*16 + (k&15).
__device__ inline v16h ldsB(const _Float16* base, int lane) {
  return *(const v16h*)(base + lane * 16);
}

__device__ inline int swzB(int k, int n) {
  return ((k >> 4) << 8) + (n << 4) + (k & 15);
}

// ------------- prep kernels: one-time weight conversion to swizzled f16 -------------
__global__ void prep_qkv(const float* __restrict__ w, _Float16* __restrict__ o) {
  int i = blockIdx.x * 256 + threadIdx.x;
  if (i >= 3 * kC * kC) return;                    // 288 x 96
  int row = i / kC, k = i % kC;
  o[(row >> 4) * 1536 + swzB(k, row & 15)] = (_Float16)w[i];
}
__global__ void prep_out(const float* __restrict__ w, _Float16* __restrict__ o) {
  int i = blockIdx.x * 256 + threadIdx.x;
  if (i >= kC * kC) return;                        // 96 x 96
  int row = i / kC, k = i % kC;
  o[(row >> 4) * 1536 + swzB(k, row & 15)] = (_Float16)w[i];
}
__global__ void prep_ff1(const float* __restrict__ w, _Float16* __restrict__ o) {
  int i = blockIdx.x * 256 + threadIdx.x;
  if (i >= kFF * kC) return;                       // 2048 x 96
  int f = i / kC, k = i % kC;
  int chunk = f >> 5, zi = (f >> 4) & 1, n = f & 15;
  o[chunk * 3072 + zi * 1536 + swzB(k, n)] = (_Float16)w[i];
}
__global__ void prep_ff2(const float* __restrict__ w, _Float16* __restrict__ o) {
  int i = blockIdx.x * 256 + threadIdx.x;
  if (i >= kC * kFF) return;                       // 96 x 2048
  int n = i / kFF, kg = i % kFF;
  int chunk = kg >> 5, k = kg & 31;
  o[chunk * 3072 + (n >> 4) * 512 + swzB(k, n & 15)] = (_Float16)w[i];
}

// ------------------- Kernel 1: transpose + QKV projection -------------------
__global__ void k_qkv(const float* __restrict__ x,
                      const _Float16* __restrict__ wsw,   // pre-swizzled 18*1536
                      const float* __restrict__ b_qkv,
                      float* __restrict__ h_f32,
                      _Float16* __restrict__ qkv) {
  __shared__ alignas(32) _Float16 As[16 * kC];
  __shared__ alignas(32) _Float16 Bs[18 * 1536];   // full QKV weight image, 54KB
  int lane = threadIdx.x;
  int b  = blockIdx.x >> 7;
  int t0 = (blockIdx.x & 127) << 4;

  for (int u = lane; u < (18 * 1536) / 8; u += 32)
    copy16(Bs + u * 8, wsw + u * 8);

  for (int idx = lane; idx < 16 * kC; idx += 32) {
    int c = idx >> 4, m = idx & 15;
    float v = x[((size_t)b * kC + c) * kT + (t0 + m)];
    As[m * kC + c] = (_Float16)v;
    h_f32[((size_t)(b * kT + t0 + m)) * kC + c] = v;
  }
  wait_async();
  __syncthreads();

  v16h aF[3];
#pragma unroll
  for (int kc = 0; kc < 3; ++kc) aF[kc] = ldsA(As + kc * 32, kC, lane);

  int n = lane & 15, hs = lane >> 4;
  for (int nt = 0; nt < 18; ++nt) {
    v8f acc = {};
#pragma unroll
    for (int kc = 0; kc < 3; ++kc)
      acc = wmma16(aF[kc], ldsB(Bs + nt * 1536 + kc * 512, lane), acc);
    float bias = b_qkv[nt * 16 + n];
#pragma unroll
    for (int r = 0; r < 8; ++r) {
      int t = t0 + r + hs * 8;
      qkv[((size_t)(b * kT + t)) * (3 * kC) + nt * 16 + n] =
          (_Float16)(acc[r] + bias);
    }
  }
}

// ------------------- Kernel 2: banded attention -------------------
__global__ void k_attn(const _Float16* __restrict__ qkv,
                       _Float16* __restrict__ ctx) {
  __shared__ alignas(32) _Float16 Kl[96 * kDh];
  __shared__ alignas(32) _Float16 Vt[kDh * 96];
  __shared__ alignas(32) _Float16 Ql[16 * kDh];
  __shared__ alignas(32) _Float16 Pl[3 * 512];
  int lane = threadIdx.x;
  int t0 = blockIdx.x << 4;
  int h  = blockIdx.y;
  int b  = blockIdx.z;
  int s0 = t0 - 32;

  for (int idx = lane; idx < (96 * kDh) / 2; idx += 32) {
    int lin = idx * 2;
    int row = lin >> 5, d = lin & 31;
    int s = s0 + row;
    v2h kv = {(_Float16)0.f, (_Float16)0.f};
    v2h vv = kv;
    if (s >= 0 && s < kT) {
      const _Float16* p = qkv + ((size_t)(b * kT + s)) * (3 * kC) + h * kDh + d;
      kv = *(const v2h*)(p + kC);
      vv = *(const v2h*)(p + 2 * kC);
    }
    *(v2h*)&Kl[lin] = kv;
    Vt[(size_t)d * 96 + row] = vv[0];
    Vt[(size_t)(d + 1) * 96 + row] = vv[1];
  }
  for (int idx = lane; idx < (16 * kDh) / 2; idx += 32) {
    int lin = idx * 2;
    int row = lin >> 5, d = lin & 31;
    *(v2h*)&Ql[lin] = *(const v2h*)(qkv + ((size_t)(b * kT + t0 + row)) * (3 * kC) +
                                    h * kDh + d);
  }
  for (int idx = lane; idx < 128; idx += 32)
    *(v2h*)&Pl[2 * 512 + 256 + idx * 2] = (v2h){(_Float16)0.f, (_Float16)0.f};
  __syncthreads();

  int n = lane & 15, hs = lane >> 4;
  v16h bq = *(const v16h*)(Ql + n * kDh + hs * 16);

  v8f sc[5];
#pragma unroll
  for (int j = 0; j < 5; ++j) {
    v8f z = {};
    sc[j] = wmma16(ldsA(Kl + j * 16 * kDh, kDh, lane), bq, z);
  }

  int t = t0 + n;
  float mx = -__builtin_inff();
#pragma unroll
  for (int j = 0; j < 5; ++j)
#pragma unroll
    for (int r = 0; r < 8; ++r) {
      int s = s0 + j * 16 + r + hs * 8;
      float v = sc[j][r] * kScale;
      bool ok = (s >= 0) && (s < kT) && (s - t <= 32) && (t - s <= 32);
      v = ok ? v : -__builtin_inff();
      sc[j][r] = v;
      mx = fmaxf(mx, v);
    }
  mx = fmaxf(mx, __shfl_xor(mx, 16, 32));
  float sum = 0.f;
#pragma unroll
  for (int j = 0; j < 5; ++j)
#pragma unroll
    for (int r = 0; r < 8; ++r) {
      float e = __expf(sc[j][r] - mx);
      sc[j][r] = e;
      sum += e;
    }
  sum += __shfl_xor(sum, 16, 32);
  float inv = 1.f / sum;
#pragma unroll
  for (int j = 0; j < 5; ++j)
#pragma unroll
    for (int r = 0; r < 8; r += 2) {
      int key = j * 16 + r + hs * 8;
      v2h pk = {(_Float16)(sc[j][r] * inv), (_Float16)(sc[j][r + 1] * inv)};
      *(v2h*)&Pl[swzB(key, n)] = pk;
    }
  __syncthreads();

#pragma unroll
  for (int dt = 0; dt < 2; ++dt) {
    v8f acc = {};
#pragma unroll
    for (int c = 0; c < 3; ++c)
      acc = wmma16(ldsA(Vt + dt * 16 * 96 + c * 32, 96, lane),
                   ldsB(Pl + c * 512, lane), acc);
    _Float16* cp = ctx + ((size_t)(b * kT + t0 + n)) * kC + h * kDh + dt * 16 + hs * 8;
#pragma unroll
    for (int r = 0; r < 8; r += 2) {
      v2h pk = {(_Float16)acc[r], (_Float16)acc[r + 1]};
      *(v2h*)(cp + r) = pk;
    }
  }
}

// ------------------- Kernel 3: out-proj + residual + LayerNorm1 -------------------
__global__ void k_proj_ln1(const _Float16* __restrict__ ctx,
                           const _Float16* __restrict__ wsw,  // pre-swizzled 6*1536
                           const float* __restrict__ b_out,
                           const float* __restrict__ h_f32,
                           const float* __restrict__ g1,
                           const float* __restrict__ bb1,
                           float* __restrict__ h1_f32,
                           _Float16* __restrict__ h1_f16) {
  __shared__ alignas(32) _Float16 As[16 * kC];
  __shared__ alignas(32) _Float16 Bs[6 * 1536];
  int lane = threadIdx.x;
  int b  = blockIdx.x >> 7;
  int t0 = (blockIdx.x & 127) << 4;

  const _Float16* ap = ctx + ((size_t)(b * kT + t0)) * kC;
  for (int u = lane; u < (16 * kC) / 8; u += 32) copy16(As + u * 8, ap + u * 8);
  for (int u = lane; u < (6 * 1536) / 8; u += 32) copy16(Bs + u * 8, wsw + u * 8);
  wait_async();
  __syncthreads();

  v16h aF[3];
#pragma unroll
  for (int kc = 0; kc < 3; ++kc) aF[kc] = ldsA(As + kc * 32, kC, lane);

  v8f acc[6];
#pragma unroll
  for (int nt = 0; nt < 6; ++nt) {
    v8f a0 = {};
#pragma unroll
    for (int kc = 0; kc < 3; ++kc)
      a0 = wmma16(aF[kc], ldsB(Bs + nt * 1536 + kc * 512, lane), a0);
    acc[nt] = a0;
  }

  int n = lane & 15, hs = lane >> 4;
#pragma unroll
  for (int r = 0; r < 8; ++r) {
    int t = t0 + r + hs * 8;
    float v[6];
    float s = 0.f;
#pragma unroll
    for (int nt = 0; nt < 6; ++nt) {
      int col = nt * 16 + n;
      v[nt] = acc[nt][r] + b_out[col] + h_f32[((size_t)(b * kT + t)) * kC + col];
      s += v[nt];
    }
#pragma unroll
    for (int m = 1; m < 16; m <<= 1) s += __shfl_xor(s, m, 32);
    float mean = s * (1.f / kC);
    float vp = 0.f;
#pragma unroll
    for (int nt = 0; nt < 6; ++nt) { float d = v[nt] - mean; vp += d * d; }
#pragma unroll
    for (int m = 1; m < 16; m <<= 1) vp += __shfl_xor(vp, m, 32);
    float rstd = rsqrtf(vp * (1.f / kC) + kEps);
#pragma unroll
    for (int nt = 0; nt < 6; ++nt) {
      int col = nt * 16 + n;
      float o = (v[nt] - mean) * rstd * g1[col] + bb1[col];
      size_t oi = ((size_t)(b * kT + t)) * kC + col;
      h1_f32[oi] = o;
      h1_f16[oi] = (_Float16)o;
    }
  }
}

// ------------------- Kernel 4: fused FF + residual + LN2 + LNf + transpose -------------------
__global__ void k_ff(const _Float16* __restrict__ h1_f16,
                     const float* __restrict__ h1_f32,
                     const _Float16* __restrict__ w1sw,  // 64 chunks * 3072
                     const float* __restrict__ b_ff1,
                     const _Float16* __restrict__ w2sw,  // 64 chunks * 3072
                     const float* __restrict__ b_ff2,
                     const float* __restrict__ g2, const float* __restrict__ bb2,
                     const float* __restrict__ gf, const float* __restrict__ bbf,
                     float* __restrict__ out) {
  __shared__ alignas(32) _Float16 As[16 * kC];
  __shared__ alignas(32) _Float16 B1[2 * 1536];
  __shared__ alignas(32) _Float16 B2[6 * 512];
  __shared__ alignas(32) _Float16 Zs[16 * 32];
  int lane = threadIdx.x;
  int b  = blockIdx.x >> 7;
  int t0 = (blockIdx.x & 127) << 4;
  int n = lane & 15, hs = lane >> 4;

  const _Float16* ap = h1_f16 + ((size_t)(b * kT + t0)) * kC;
  for (int u = lane; u < (16 * kC) / 8; u += 32) copy16(As + u * 8, ap + u * 8);
  wait_async();
  __syncthreads();

  v16h aF[3];
#pragma unroll
  for (int kc = 0; kc < 3; ++kc) aF[kc] = ldsA(As + kc * 32, kC, lane);

  v8f acc[6];
#pragma unroll
  for (int nt = 0; nt < 6; ++nt) acc[nt] = (v8f){};

  for (int f0 = 0; f0 < kFF; f0 += 32) {
    const _Float16* g1p = w1sw + (size_t)(f0 >> 5) * 3072;
    const _Float16* g2p = w2sw + (size_t)(f0 >> 5) * 3072;
    for (int u = lane; u < 3072 / 8; u += 32) {
      copy16(B1 + u * 8, g1p + u * 8);
      copy16(B2 + u * 8, g2p + u * 8);
    }
    wait_async();
    __syncthreads();

#pragma unroll
    for (int zi = 0; zi < 2; ++zi) {
      v8f z = {};
#pragma unroll
      for (int kc = 0; kc < 3; ++kc)
        z = wmma16(aF[kc], ldsB(B1 + zi * 1536 + kc * 512, lane), z);
      float bias = b_ff1[f0 + zi * 16 + n];
#pragma unroll
      for (int r = 0; r < 8; ++r) {
        float zz = fmaxf(z[r] + bias, 0.f);
        Zs[(r + hs * 8) * 32 + zi * 16 + n] = (_Float16)zz;
      }
    }
    __syncthreads();

    v16h a2 = ldsA(Zs, 32, lane);
#pragma unroll
    for (int nt = 0; nt < 6; ++nt)
      acc[nt] = wmma16(a2, ldsB(B2 + nt * 512, lane), acc[nt]);
    __syncthreads();
  }

#pragma unroll
  for (int r = 0; r < 8; ++r) {
    int t = t0 + r + hs * 8;
    float v[6];
    float s = 0.f;
#pragma unroll
    for (int nt = 0; nt < 6; ++nt) {
      int col = nt * 16 + n;
      v[nt] = acc[nt][r] + b_ff2[col] + h1_f32[((size_t)(b * kT + t)) * kC + col];
      s += v[nt];
    }
#pragma unroll
    for (int m = 1; m < 16; m <<= 1) s += __shfl_xor(s, m, 32);
    float mean = s * (1.f / kC);
    float vp = 0.f;
#pragma unroll
    for (int nt = 0; nt < 6; ++nt) { float d = v[nt] - mean; vp += d * d; }
#pragma unroll
    for (int m = 1; m < 16; m <<= 1) vp += __shfl_xor(vp, m, 32);
    float rstd = rsqrtf(vp * (1.f / kC) + kEps);

    float u[6];
    float s2 = 0.f;
#pragma unroll
    for (int nt = 0; nt < 6; ++nt) {
      int col = nt * 16 + n;
      u[nt] = (v[nt] - mean) * rstd * g2[col] + bb2[col];
      s2 += u[nt];
    }
#pragma unroll
    for (int m = 1; m < 16; m <<= 1) s2 += __shfl_xor(s2, m, 32);
    float mean2 = s2 * (1.f / kC);
    float vp2 = 0.f;
#pragma unroll
    for (int nt = 0; nt < 6; ++nt) { float d = u[nt] - mean2; vp2 += d * d; }
#pragma unroll
    for (int m = 1; m < 16; m <<= 1) vp2 += __shfl_xor(vp2, m, 32);
    float rstd2 = rsqrtf(vp2 * (1.f / kC) + kEps);
#pragma unroll
    for (int nt = 0; nt < 6; ++nt) {
      int col = nt * 16 + n;
      float o = (u[nt] - mean2) * rstd2 * gf[col] + bbf[col];
      out[((size_t)b * kC + col) * kT + t] = o;
    }
  }
}

extern "C" void kernel_launch(void* const* d_in, const int* in_sizes, int n_in,
                              void* d_out, int out_size, void* d_ws, size_t ws_size,
                              hipStream_t stream) {
  (void)in_sizes; (void)n_in; (void)out_size; (void)ws_size;
  const float* x      = (const float*)d_in[0];
  const float* w_qkv  = (const float*)d_in[1];
  const float* b_qkv  = (const float*)d_in[2];
  const float* w_out  = (const float*)d_in[3];
  const float* b_out  = (const float*)d_in[4];
  const float* ln1_g  = (const float*)d_in[5];
  const float* ln1_b  = (const float*)d_in[6];
  const float* w_ff1  = (const float*)d_in[7];
  const float* b_ff1  = (const float*)d_in[8];
  const float* w_ff2  = (const float*)d_in[9];
  const float* b_ff2  = (const float*)d_in[10];
  const float* ln2_g  = (const float*)d_in[11];
  const float* ln2_b  = (const float*)d_in[12];
  const float* lnf_g  = (const float*)d_in[13];
  const float* lnf_b  = (const float*)d_in[14];
  float* out = (float*)d_out;

  char* ws = (char*)d_ws;
  constexpr size_t szHf32 = (size_t)kB * kT * kC * sizeof(float);
  constexpr size_t szQkv  = (size_t)kB * kT * 3 * kC * sizeof(_Float16);
  constexpr size_t szCtx  = (size_t)kB * kT * kC * sizeof(_Float16);
  constexpr size_t szH1f  = szHf32;
  constexpr size_t szH1h  = szCtx;
  size_t off = 0;
  float*    h_f32  = (float*)(ws + off);    off += szHf32;
  _Float16* qkv    = (_Float16*)(ws + off); off += szQkv;
  _Float16* ctx    = (_Float16*)(ws + off); off += szCtx;
  float*    h1_f32 = (float*)(ws + off);    off += szH1f;
  _Float16* h1_f16 = (_Float16*)(ws + off); off += szH1h;
  _Float16* wqkv_sw = (_Float16*)(ws + off); off += (size_t)18 * 1536 * 2;
  _Float16* wout_sw = (_Float16*)(ws + off); off += (size_t)6 * 1536 * 2;
  _Float16* wff1_sw = (_Float16*)(ws + off); off += (size_t)64 * 3072 * 2;
  _Float16* wff2_sw = (_Float16*)(ws + off); off += (size_t)64 * 3072 * 2;

  dim3 blk(32, 1, 1);
  prep_qkv<<<dim3((3 * kC * kC + 255) / 256), 256, 0, stream>>>(w_qkv, wqkv_sw);
  prep_out<<<dim3((kC * kC + 255) / 256), 256, 0, stream>>>(w_out, wout_sw);
  prep_ff1<<<dim3((kFF * kC + 255) / 256), 256, 0, stream>>>(w_ff1, wff1_sw);
  prep_ff2<<<dim3((kC * kFF + 255) / 256), 256, 0, stream>>>(w_ff2, wff2_sw);

  k_qkv<<<dim3(kB * kTilesT), blk, 0, stream>>>(x, wqkv_sw, b_qkv, h_f32, qkv);
  k_attn<<<dim3(kTilesT, kH, kB), blk, 0, stream>>>(qkv, ctx);
  k_proj_ln1<<<dim3(kB * kTilesT), blk, 0, stream>>>(ctx, wout_sw, b_out, h_f32,
                                                     ln1_g, ln1_b, h1_f32, h1_f16);
  k_ff<<<dim3(kB * kTilesT), blk, 0, stream>>>(h1_f16, h1_f32, wff1_sw, b_ff1,
                                               wff2_sw, b_ff2, ln2_g, ln2_b,
                                               lnf_g, lnf_b, out);
}
